// Net_upsampling_9758165697037
// MI455X (gfx1250) — matrix operand order repaired and checked by reading, not compiled, not run on hardware
//
#include <hip/hip_runtime.h>
#include <hip/hip_bf16.h>
#include <stdint.h>

// =====================================================================
// MI455X (gfx1250) implementation.
//
// Strategy (from MI455X roofline):
//  * All 2D ResNet convs -> im2col (f32->f16, K padded to 32) + WMMA GEMM
//    using v_wmma_f32_16x16x32_f16 (f16 inputs, f32 accum): the CDNA5
//    matrix path carries the dominant ~4 GFLOP of conv work.
//  * GEMM tiles are staged global->LDS with CDNA5 async copies
//    (global_load_async_to_lds_b128 + s_wait_asynccnt), then fragments
//    are assembled with ds_load_b128 per the ISA 16-bit A/B layouts.
//    Each wave register-blocks a 16x64 C strip (4 WMMAs per K step).
//  * 3D transposed convs / 1x1x1 convs (~2 GMAC total) -> direct VALU
//    kernels (too small to justify matrix-path data movement).
//  * grid_sample + point gather -> bandwidth-bound gather kernels
//    (~0.5 GB traffic vs 23.3 TB/s HBM -> microseconds).
//  * Training-mode BN: per-channel block reduction + fused
//    normalize(+residual)(+relu).
// =====================================================================

typedef __attribute__((ext_vector_type(16))) _Float16 v16h;
typedef __attribute__((ext_vector_type(8)))  float    v8f;

#define DIV_UP(a, b) (((a) + (b) - 1) / (b))

// CDNA5 async global->LDS copy (ASYNCcnt-tracked), 16 bytes per lane.
__device__ __forceinline__ void async_copy16(uint32_t lds_off,
                                             const void* gptr) {
  asm volatile("global_load_async_to_lds_b128 %0, %1, off"
               :: "v"(lds_off), "v"(gptr) : "memory");
}
__device__ __forceinline__ void wait_async() {
  asm volatile("s_wait_asynccnt 0x0" ::: "memory");
}

// ---------------------------------------------------------------------
// WMMA GEMM:  C[Mp x N] = A[Mp x K] * Bt[N x K]^T + bias
// A row-major f16 (im2col, zero padded), Bt row-major f16 (weights in
// natural (Cout, K) layout, zero padded K), C row-major f32.
// Mp % 64 == 0, N % 64 == 0, K % 32 == 0.
// Block = 128 threads = 4 waves; block tile 64x64; each wave computes a
// 16x64 C strip -> 4 v_wmma_f32_16x16x32_f16 per K step, one shared A
// fragment. A/B tiles (64x32 f16 each) staged via async copies.
// ---------------------------------------------------------------------
__global__ __launch_bounds__(128) void gemm_wmma_kernel(
    const _Float16* __restrict__ A, const _Float16* __restrict__ Bt,
    const float* __restrict__ bias, float* __restrict__ C,
    int Mp, int N, int K)
{
  __shared__ uint32_t sA[1024];   // 64 rows x 16 dwords (K pairs)
  __shared__ uint32_t sB[1024];

  const int t = threadIdx.x;
  const int wave = t >> 5;
  const int lane = t & 31;
  const int ntb = N >> 6;
  const int bm = blockIdx.x / ntb;
  const int bn = blockIdx.x % ntb;
  const int m0 = bm << 6, n0 = bn << 6;

  const int am = lane & 15;     // A row / B row / C column within subtile
  const int kh = lane >> 4;     // K-half select
  const int Kw = K >> 1;        // K in dword pairs

  const uint32_t* gA = (const uint32_t*)A + (size_t)m0 * Kw;
  const uint32_t* gB = (const uint32_t*)Bt + (size_t)n0 * Kw;

  // Per-thread staging: two 16B chunks for A, two for B per K step.
  const int c4 = (t & 3) * 4;   // dword column within a 16-dword tile row
  const int r0 = t >> 2;        // tile row 0..31 (second chunk: +32)
  const uint32_t ldsA0 = (uint32_t)(uintptr_t)&sA[r0 * 16 + c4];
  const uint32_t ldsA1 = (uint32_t)(uintptr_t)&sA[(r0 + 32) * 16 + c4];
  const uint32_t ldsB0 = (uint32_t)(uintptr_t)&sB[r0 * 16 + c4];
  const uint32_t ldsB1 = (uint32_t)(uintptr_t)&sB[(r0 + 32) * 16 + c4];

  v8f acc0 = {}, acc1 = {}, acc2 = {}, acc3 = {};

#pragma unroll 2
  for (int k0 = 0; k0 < Kw; k0 += 16) {
    async_copy16(ldsA0, gA + (size_t)r0 * Kw + k0 + c4);
    async_copy16(ldsA1, gA + (size_t)(r0 + 32) * Kw + k0 + c4);
    async_copy16(ldsB0, gB + (size_t)r0 * Kw + k0 + c4);
    async_copy16(ldsB1, gB + (size_t)(r0 + 32) * Kw + k0 + c4);
    wait_async();
    __syncthreads();

    union { uint32_t u[8]; v16h h; } fa, fb0, fb1, fb2, fb3;
#pragma unroll
    for (int v = 0; v < 8; ++v) {
      // A 16x32 f16 fragment: VGPR v<4 -> K pair v + 4*khalf,
      //                       v>=4    -> K pair 8 + (v-4) + 4*khalf
      int pa = (v < 4) ? (v + 4 * kh) : (8 + (v - 4) + 4 * kh);
      fa.u[v] = sA[(wave * 16 + am) * 16 + pa];
      // B 32x16 f16 fragment: same K-pair order as A
      int pb = (v < 4) ? (v + 4 * kh) : (8 + (v - 4) + 4 * kh);
      fb0.u[v] = sB[am * 16 + pb];
      fb1.u[v] = sB[(16 + am) * 16 + pb];
      fb2.u[v] = sB[(32 + am) * 16 + pb];
      fb3.u[v] = sB[(48 + am) * 16 + pb];
    }
    acc0 = __builtin_amdgcn_wmma_f32_16x16x32_f16(
        false, fa.h, false, fb0.h, (short)0, acc0, false, false);
    acc1 = __builtin_amdgcn_wmma_f32_16x16x32_f16(
        false, fa.h, false, fb1.h, (short)0, acc1, false, false);
    acc2 = __builtin_amdgcn_wmma_f32_16x16x32_f16(
        false, fa.h, false, fb2.h, (short)0, acc2, false, false);
    acc3 = __builtin_amdgcn_wmma_f32_16x16x32_f16(
        false, fa.h, false, fb3.h, (short)0, acc3, false, false);
    __syncthreads();
  }

  const int mrow = m0 + wave * 16;
  const float b0 = bias[n0 + am];
  const float b1 = bias[n0 + 16 + am];
  const float b2 = bias[n0 + 32 + am];
  const float b3 = bias[n0 + 48 + am];
#pragma unroll
  for (int r = 0; r < 8; ++r) {
    int m = mrow + r + 8 * kh;    // C: VGPR r -> row r (+8 for upper half)
    float* crow = C + (size_t)m * N + n0 + am;
    crow[0]  = acc0[r] + b0;
    crow[16] = acc1[r] + b1;
    crow[32] = acc2[r] + b2;
    crow[48] = acc3[r] + b3;
  }
}

// ---------------------------------------------------------------------
// Weight pack: (Cout, Kv) f32 -> (Cout, Kp) f16, zero padded.
// ---------------------------------------------------------------------
__global__ void pack_w_kernel(const float* __restrict__ w,
                              _Float16* __restrict__ out,
                              int Cout, int Kv, int Kp)
{
  size_t idx = (size_t)blockIdx.x * blockDim.x + threadIdx.x;
  size_t tot = (size_t)Cout * Kp;
  if (idx >= tot) return;
  int co = (int)(idx / Kp), k = (int)(idx % Kp);
  float v = (k < Kv) ? w[(size_t)co * Kv + k] : 0.f;
  out[idx] = (_Float16)v;
}

// ---------------------------------------------------------------------
// im2col for NHWC input (B=4). A[m, k] with m=(b,oh,ow), k=ci*KH*KW+kh*KW+kw.
// Zero padded to (Mp, Kp).
// ---------------------------------------------------------------------
__global__ void im2col2d_kernel(const float* __restrict__ in,
                                _Float16* __restrict__ A,
                                int H, int W, int Cin, int KK, int stride,
                                int pad, int OH, int OW, int Kp, int Mp)
{
  size_t idx = (size_t)blockIdx.x * blockDim.x + threadIdx.x;
  size_t tot = (size_t)Mp * Kp;
  if (idx >= tot) return;
  int m = (int)(idx / Kp), k = (int)(idx % Kp);
  int M = 4 * OH * OW;
  int Kv = Cin * KK * KK;
  float v = 0.f;
  if (m < M && k < Kv) {
    int b = m / (OH * OW), rem = m % (OH * OW);
    int oh = rem / OW, ow = rem % OW;
    int ci = k / (KK * KK), r2 = k % (KK * KK);
    int khh = r2 / KK, kww = r2 % KK;
    int ih = oh * stride - pad + khh;
    int iw = ow * stride - pad + kww;
    if (ih >= 0 && ih < H && iw >= 0 && iw < W)
      v = in[(((size_t)b * H + ih) * W + iw) * Cin + ci];
  }
  A[idx] = (_Float16)v;
}

// ---------------------------------------------------------------------
// Training-mode BatchNorm over NHWC (M x C): stats + fused apply.
// ---------------------------------------------------------------------
__global__ __launch_bounds__(256) void bn_stats_mc_kernel(
    const float* __restrict__ x, int M, int C,
    float* __restrict__ mean, float* __restrict__ rstd)
{
  int c = blockIdx.x;
  float s = 0.f, s2 = 0.f;
  for (int m = threadIdx.x; m < M; m += 256) {
    float v = x[(size_t)m * C + c];
    s += v; s2 += v * v;
  }
  __shared__ float rs[256], rq[256];
  rs[threadIdx.x] = s; rq[threadIdx.x] = s2;
  __syncthreads();
  for (int o = 128; o > 0; o >>= 1) {
    if (threadIdx.x < o) {
      rs[threadIdx.x] += rs[threadIdx.x + o];
      rq[threadIdx.x] += rq[threadIdx.x + o];
    }
    __syncthreads();
  }
  if (threadIdx.x == 0) {
    float mu = rs[0] / (float)M;
    float var = rq[0] / (float)M - mu * mu;
    mean[c] = mu;
    rstd[c] = rsqrtf(var + 1e-5f);
  }
}

__global__ void bn_apply_mc_kernel(const float* __restrict__ x,
                                   const float* __restrict__ res,
                                   float* __restrict__ y,
                                   const float* __restrict__ mean,
                                   const float* __restrict__ rstd,
                                   const float* __restrict__ g,
                                   const float* __restrict__ b,
                                   size_t M, int C, int relu)
{
  size_t idx = (size_t)blockIdx.x * blockDim.x + threadIdx.x;
  size_t tot = M * (size_t)C;
  if (idx >= tot) return;
  int c = (int)(idx % C);
  float v = (x[idx] - mean[c]) * rstd[c] * g[c] + b[c];
  if (res) v += res[idx];
  if (relu) v = fmaxf(v, 0.f);
  y[idx] = v;
}

// ---------------------------------------------------------------------
// Training-mode BatchNorm over NCS layout (B=4, C channels, S spatial).
// ---------------------------------------------------------------------
__global__ __launch_bounds__(256) void bn_stats_ncs_kernel(
    const float* __restrict__ x, int C, int S,
    float* __restrict__ mean, float* __restrict__ rstd)
{
  int c = blockIdx.x;
  long T = 4L * S;
  float s = 0.f, s2 = 0.f;
  for (long t = threadIdx.x; t < T; t += 256) {
    long n = t / S, sp = t % S;
    float v = x[((size_t)n * C + c) * S + sp];
    s += v; s2 += v * v;
  }
  __shared__ float rs[256], rq[256];
  rs[threadIdx.x] = s; rq[threadIdx.x] = s2;
  __syncthreads();
  for (int o = 128; o > 0; o >>= 1) {
    if (threadIdx.x < o) {
      rs[threadIdx.x] += rs[threadIdx.x + o];
      rq[threadIdx.x] += rq[threadIdx.x + o];
    }
    __syncthreads();
  }
  if (threadIdx.x == 0) {
    float mu = rs[0] / (float)T;
    float var = rq[0] / (float)T - mu * mu;
    mean[c] = mu;
    rstd[c] = rsqrtf(var + 1e-5f);
  }
}

__global__ void bn_apply_ncs_kernel(float* __restrict__ xy,
                                    const float* __restrict__ mean,
                                    const float* __restrict__ rstd,
                                    const float* __restrict__ g,
                                    const float* __restrict__ b,
                                    int C, int S, int relu)
{
  size_t idx = (size_t)blockIdx.x * blockDim.x + threadIdx.x;
  size_t tot = 4ull * C * S;
  if (idx >= tot) return;
  int c = (int)((idx / S) % C);
  float v = (xy[idx] - mean[c]) * rstd[c] * g[c] + b[c];
  if (relu) v = fmaxf(v, 0.f);
  xy[idx] = v;
}

// ---------------------------------------------------------------------
// MaxPool 3x3 s2 p1, NHWC, B=4.
// ---------------------------------------------------------------------
__global__ void maxpool_nhwc_kernel(const float* __restrict__ in,
                                    float* __restrict__ out,
                                    int H, int W, int C, int OH, int OW)
{
  size_t idx = (size_t)blockIdx.x * blockDim.x + threadIdx.x;
  size_t tot = 4ull * OH * OW * C;
  if (idx >= tot) return;
  int c = (int)(idx % C);
  size_t r = idx / C;
  int ow = (int)(r % OW); r /= OW;
  int oh = (int)(r % OH);
  int b = (int)(r / OH);
  float m = -3.4e38f;
  for (int khh = 0; khh < 3; ++khh) {
    int ih = oh * 2 - 1 + khh;
    if (ih < 0 || ih >= H) continue;
    for (int kww = 0; kww < 3; ++kww) {
      int iw = ow * 2 - 1 + kww;
      if (iw < 0 || iw >= W) continue;
      m = fmaxf(m, in[(((size_t)b * H + ih) * W + iw) * C + c]);
    }
  }
  out[idx] = m;
}

// AdaptiveAvgPool2d(1): NHWC (4,4,4,512) -> (4,512) == NCDHW (4,8,4,4,4)
__global__ void avgpool_kernel(const float* __restrict__ in,
                               float* __restrict__ out)
{
  int idx = blockIdx.x * blockDim.x + threadIdx.x;
  if (idx >= 4 * 512) return;
  int b = idx >> 9, c = idx & 511;
  float s = 0.f;
  for (int p = 0; p < 16; ++p) s += in[((size_t)b * 16 + p) * 512 + c];
  out[idx] = s * (1.f / 16.f);
}

// ---------------------------------------------------------------------
// ConvTranspose3d, k=4, stride=2, NCDHW, B=4. w: (Cin, Cout, 4,4,4).
// Optional fused sigmoid for the final layer.
// ---------------------------------------------------------------------
__global__ void convt3d_kernel(const float* __restrict__ in,
                               const float* __restrict__ w,
                               const float* __restrict__ bias,
                               float* __restrict__ out,
                               int Cin, int ID, int IH, int IW,
                               int Cout, int OD, int OH, int OW,
                               int pd, int ph, int pw, int sigm)
{
  size_t idx = (size_t)blockIdx.x * blockDim.x + threadIdx.x;
  size_t tot = 4ull * Cout * OD * OH * OW;
  if (idx >= tot) return;
  int ow = (int)(idx % OW);
  size_t r = idx / OW;
  int oh = (int)(r % OH); r /= OH;
  int od = (int)(r % OD); r /= OD;
  int co = (int)(r % Cout);
  int b = (int)(r / Cout);

  float acc = bias[co];
  const size_t ivol = (size_t)ID * IH * IW;
  for (int kd = 0; kd < 4; ++kd) {
    int tt = od + pd - kd;
    if (tt < 0 || (tt & 1)) continue;
    int id = tt >> 1; if (id >= ID) continue;
    for (int khh = 0; khh < 4; ++khh) {
      int u = oh + ph - khh;
      if (u < 0 || (u & 1)) continue;
      int ih = u >> 1; if (ih >= IH) continue;
      for (int kww = 0; kww < 4; ++kww) {
        int q = ow + pw - kww;
        if (q < 0 || (q & 1)) continue;
        int iw = q >> 1; if (iw >= IW) continue;
        const float* ip = in + (size_t)b * Cin * ivol
                          + ((size_t)id * IH + ih) * IW + iw;
        int wk = kd * 16 + khh * 4 + kww;
        for (int ci = 0; ci < Cin; ++ci)
          acc += ip[(size_t)ci * ivol] * w[((size_t)ci * Cout + co) * 64 + wk];
      }
    }
  }
  if (sigm) acc = 1.f / (1.f + expf(-acc));
  out[idx] = acc;
}

// 1x1x1 Conv3d, NCDHW, B=4. w: (Cout, Cin).
__global__ void conv3d1_kernel(const float* __restrict__ in,
                               const float* __restrict__ w,
                               const float* __restrict__ bias,
                               float* __restrict__ out,
                               int Cin, int Cout, int S)
{
  size_t idx = (size_t)blockIdx.x * blockDim.x + threadIdx.x;
  size_t tot = 4ull * Cout * S;
  if (idx >= tot) return;
  int s = (int)(idx % S);
  size_t r = idx / S;
  int co = (int)(r % Cout);
  int b = (int)(r / Cout);
  float acc = bias[co];
  for (int ci = 0; ci < Cin; ++ci)
    acc += in[((size_t)b * Cin + ci) * S + s] * w[(size_t)co * Cin + ci];
  out[idx] = acc;
}

// ---------------------------------------------------------------------
// grid_sample 3D, bilinear, zeros padding, align_corners=False.
// vol (4,1,D,H,W); grid (150^3,3) shared across batch; out (4,150^3).
// ---------------------------------------------------------------------
__global__ void grid_sample_kernel(const float* __restrict__ vol,
                                   const float* __restrict__ grid,
                                   float* __restrict__ out,
                                   int D, int H, int W)
{
  const size_t G = 150ull * 150ull * 150ull;
  size_t idx = (size_t)blockIdx.x * blockDim.x + threadIdx.x;
  if (idx >= 4 * G) return;
  int b = (int)(idx / G);
  size_t gI = idx % G;
  float gx = grid[gI * 3 + 0];
  float gy = grid[gI * 3 + 1];
  float gz = grid[gI * 3 + 2];
  float ix = ((gx + 1.f) * W - 1.f) * 0.5f;
  float iy = ((gy + 1.f) * H - 1.f) * 0.5f;
  float iz = ((gz + 1.f) * D - 1.f) * 0.5f;
  float x0 = floorf(ix), y0 = floorf(iy), z0 = floorf(iz);
  const float* vb = vol + (size_t)b * D * H * W;
  float o = 0.f;
  for (int dz = 0; dz < 2; ++dz)
    for (int dy = 0; dy < 2; ++dy)
      for (int dx = 0; dx < 2; ++dx) {
        float xc = x0 + dx, yc = y0 + dy, zc = z0 + dz;
        float wgt = (1.f - fabsf(ix - xc)) * (1.f - fabsf(iy - yc)) *
                    (1.f - fabsf(iz - zc));
        if (xc >= 0.f && xc < (float)W && yc >= 0.f && yc < (float)H &&
            zc >= 0.f && zc < (float)D) {
          int xi = (int)xc, yi = (int)yc, zi = (int)zc;
          o += vb[((size_t)zi * H + yi) * W + xi] * wgt;
        }
      }
  out[idx] = o;
}

// Final masked gather: x (4,262144,3); vol2 (4,150,150,150); out (4*262144,2)
__global__ void gather_out_kernel(const float* __restrict__ x,
                                  const float* __restrict__ vol2,
                                  float* __restrict__ out)
{
  const int N = 262144;
  size_t idx = (size_t)blockIdx.x * blockDim.x + threadIdx.x;
  if (idx >= 4ull * N) return;
  int b = (int)(idx / N);
  float vx = x[idx * 3 + 0], vy = x[idx * 3 + 1], vz = x[idx * 3 + 2];
  bool inside = (vx >= 0.f) && (vx < 150.f) && (vy >= 0.f) && (vy < 150.f);
  int xi = min(max((int)vx, 0), 149);
  int yi = min(max((int)vy, 0), 149);
  int zi = min(max((int)vz, 0), 149);
  float yp = inside
                 ? vol2[(((size_t)b * 150 + xi) * 150 + yi) * 150 + zi]
                 : 0.f;
  out[idx * 2 + 0] = 1.f - yp;
  out[idx * 2 + 1] = yp;
}

// =====================================================================
// Host orchestration
// =====================================================================
extern "C" void kernel_launch(void* const* d_in, const int* in_sizes, int n_in,
                              void* d_out, int out_size, void* d_ws,
                              size_t ws_size, hipStream_t stream)
{
  (void)in_sizes; (void)n_in; (void)out_size; (void)ws_size;

  const float* xpts  = (const float*)d_in[0];  // (4,262144,3)
  const float* image = (const float*)d_in[1];  // (4,1,128,128) == NHWC C=1
  const float* grid  = (const float*)d_in[2];  // (150,150,150,3)
  float* out = (float*)d_out;

  auto F = [&](int i) { return (const float*)d_in[i]; };

  // params leaves in JAX sorted-key tree-flatten order, starting at d_in[3]
  int pi = 3;
  const float *bn0_3d_b = F(pi++), *bn0_3d_g = F(pi++);
  const float *bn1_b = F(pi++),    *bn1_g = F(pi++);
  const float *bn2_b = F(pi++),    *bn2_g = F(pi++);
  const float *bn3_b = F(pi++),    *bn3_g = F(pi++);
  const float *bnc1_b = F(pi++),   *bnc1_g = F(pi++);
  const float *bnc2_b = F(pi++),   *bnc2_g = F(pi++);
  const float *bnc3_b = F(pi++),   *bnc3_g = F(pi++);
  const float *c1_b = F(pi++), *c1_w = F(pi++);
  const float *c2_b = F(pi++), *c2_w = F(pi++);
  const float *c3_b = F(pi++), *c3_w = F(pi++);
  const float *cnn_bn0_b = F(pi++), *cnn_bn0_g = F(pi++);
  const float *cnn_c0_b = F(pi++),  *cnn_c0_w = F(pi++);
  // (ResNet block params consumed in order inside the layer loop; ct* after.)

  // ---- workspace layout ----
  const size_t MBy = 1024ull * 1024ull;
  char* ws = (char*)d_ws;
  _Float16* Abuf = (_Float16*)(ws + 0);            // im2col   (8 MB)
  _Float16* Wh   = (_Float16*)(ws + 8 * MBy);      // f16 wts  (8 MB)
  float* buf0 = (float*)(ws + 16 * MBy);           // act      (4 MB)
  float* buf1 = (float*)(ws + 20 * MBy);           // act      (4 MB)
  float* buf2 = (float*)(ws + 24 * MBy);           // shortcut (4 MB)
  float* buf3 = (float*)(ws + 28 * MBy);           // act      (4 MB)
  float* bmean = (float*)(ws + 32 * MBy);
  float* brstd = (float*)(ws + 32 * MBy + 16 * 1024);
  float* t3a = (float*)(ws + 33 * MBy);            // 3D ping (20 MB)
  float* t3b = (float*)(ws + 53 * MBy);            // 3D pong (20 MB)
  float* vol = (float*)(ws + 73 * MBy);            // 4*150^3 (54 MB)

  // ---- helpers ----
  auto conv2d = [&](const float* in, int H, int W, int Cin, const float* w,
                    const float* bias, int Cout, int KK, int stride, int pad,
                    float* dst, int& OH, int& OW) {
    OH = (H + 2 * pad - KK) / stride + 1;
    OW = (W + 2 * pad - KK) / stride + 1;
    int Kv = Cin * KK * KK;
    int Kp = ((Kv + 31) / 32) * 32;
    int M = 4 * OH * OW;               // always a multiple of 64 here
    int Mp = ((M + 63) / 64) * 64;
    {
      size_t tot = (size_t)Cout * Kp;
      pack_w_kernel<<<DIV_UP(tot, 256), 256, 0, stream>>>(w, Wh, Cout, Kv, Kp);
    }
    {
      size_t tot = (size_t)Mp * Kp;
      im2col2d_kernel<<<DIV_UP(tot, 256), 256, 0, stream>>>(
          in, Abuf, H, W, Cin, KK, stride, pad, OH, OW, Kp, Mp);
    }
    {
      int blocks = (Mp / 64) * (Cout / 64);
      gemm_wmma_kernel<<<blocks, 128, 0, stream>>>(
          Abuf, Wh, bias, dst, Mp, Cout, Kp);
    }
  };

  auto bn2d = [&](const float* xin, const float* res, float* y,
                  const float* g, const float* b, int M, int C, int relu) {
    bn_stats_mc_kernel<<<C, 256, 0, stream>>>(xin, M, C, bmean, brstd);
    size_t tot = (size_t)M * C;
    bn_apply_mc_kernel<<<DIV_UP(tot, 256), 256, 0, stream>>>(
        xin, res, y, bmean, brstd, g, b, (size_t)M, C, relu);
  };

  auto bn3d = [&](float* xy, const float* g, const float* b, int C, int S,
                  int relu) {
    bn_stats_ncs_kernel<<<C, 256, 0, stream>>>(xy, C, S, bmean, brstd);
    size_t tot = 4ull * C * S;
    bn_apply_ncs_kernel<<<DIV_UP(tot, 256), 256, 0, stream>>>(
        xy, bmean, brstd, g, b, C, S, relu);
  };

  // ================= 2D ResNet =================
  int OH, OW;
  // conv0: 1->64, k7 s2 p3 : 128 -> 64
  conv2d(image, 128, 128, 1, cnn_c0_w, cnn_c0_b, 64, 7, 2, 3, buf0, OH, OW);
  // maxpool 3x3 s2 p1 : 64 -> 32
  {
    int H = OH, W = OW;
    OH = 32; OW = 32;
    size_t tot = 4ull * OH * OW * 64;
    maxpool_nhwc_kernel<<<DIV_UP(tot, 256), 256, 0, stream>>>(
        buf0, buf1, H, W, 64, OH, OW);
  }
  // bn0 + relu
  bn2d(buf1, nullptr, buf0, cnn_bn0_g, cnn_bn0_b, 4 * 32 * 32, 64, 1);

  float* cur = buf0;
  int H = 32, W = 32, Cc = 64;
  const int layer_blocks[4] = {3, 4, 6, 3};
  const int layer_cout[4] = {64, 128, 256, 512};

  for (int L = 0; L < 4; ++L) {
    for (int blk = 0; blk < layer_blocks[L]; ++blk) {
      int down = (L > 0 && blk == 0) ? 1 : 0;
      int cout = layer_cout[L];
      // block leaves (sorted keys): bn1(b,g) bn2(b,g) c1(b,w) c2(b,w)
      //                             [sc_bn(b,g) sc_c(b,w)]
      const float *bb1 = F(pi++), *bg1 = F(pi++);
      const float *bb2 = F(pi++), *bg2 = F(pi++);
      const float *cb1 = F(pi++), *cw1 = F(pi++);
      const float *cb2 = F(pi++), *cw2 = F(pi++);
      const float *sbb = nullptr, *sbg = nullptr, *scb = nullptr,
                  *scw = nullptr;
      if (down) { sbb = F(pi++); sbg = F(pi++); scb = F(pi++); scw = F(pi++); }

      const float* resid;
      if (down) {
        int soh, sow;
        conv2d(cur, H, W, Cc, scw, scb, cout, 1, 2, 0, buf2, soh, sow);
        bn2d(buf2, nullptr, buf2, sbg, sbb, 4 * soh * sow, cout, 0);
        resid = buf2;
      } else {
        resid = cur;
      }
      conv2d(cur, H, W, Cc, cw1, cb1, cout, 3, down ? 2 : 1, 1, buf1, OH, OW);
      bn2d(buf1, nullptr, buf1, bg1, bb1, 4 * OH * OW, cout, 1);
      conv2d(buf1, OH, OW, cout, cw2, cb2, cout, 3, 1, 1, buf3, OH, OW);
      bn2d(buf3, resid, buf0, bg2, bb2, 4 * OH * OW, cout, 1);
      cur = buf0; H = OH; W = OW; Cc = cout;
    }
  }

  // transposed-conv params (after cnn leaves)
  const float *ct1_b = F(pi++), *ct1_w = F(pi++);
  const float *ct2_b = F(pi++), *ct2_w = F(pi++);
  const float *ct3_b = F(pi++), *ct3_w = F(pi++);
  const float *ct4_b = F(pi++), *ct4_w = F(pi++);

  // avgpool (4,4,4,512) -> (4,512) == (4,8,4,4,4)
  avgpool_kernel<<<DIV_UP(4 * 512, 256), 256, 0, stream>>>(cur, t3a);

  // ================= 3D decoder =================
  bn3d(t3a, bn0_3d_g, bn0_3d_b, 8, 64, 1);

  // ct1: 8->64, (4,4,4) -> (10,10,10)
  {
    size_t tot = 4ull * 64 * 1000;
    convt3d_kernel<<<DIV_UP(tot, 256), 256, 0, stream>>>(
        t3a, ct1_w, ct1_b, t3b, 8, 4, 4, 4, 64, 10, 10, 10, 0, 0, 0, 0);
  }
  bn3d(t3b, bn1_g, bn1_b, 64, 1000, 1);
  {
    size_t tot = 4ull * 64 * 1000;
    conv3d1_kernel<<<DIV_UP(tot, 256), 256, 0, stream>>>(t3b, c1_w, c1_b, t3a,
                                                         64, 64, 1000);
  }
  bn3d(t3a, bnc1_g, bnc1_b, 64, 1000, 1);

  // ct2: 64->32, (10,10,10) -> (22,22,18), pad (0,0,2)
  {
    size_t tot = 4ull * 32 * 22 * 22 * 18;
    convt3d_kernel<<<DIV_UP(tot, 256), 256, 0, stream>>>(
        t3a, ct2_w, ct2_b, t3b, 64, 10, 10, 10, 32, 22, 22, 18, 0, 0, 2, 0);
  }
  bn3d(t3b, bn2_g, bn2_b, 32, 22 * 22 * 18, 1);
  {
    size_t tot = 4ull * 32 * 8712;
    conv3d1_kernel<<<DIV_UP(tot, 256), 256, 0, stream>>>(t3b, c2_w, c2_b, t3a,
                                                         32, 32, 8712);
  }
  bn3d(t3a, bnc2_g, bnc2_b, 32, 8712, 1);

  // ct3: 32->16, (22,22,18) -> (46,46,34), pad (0,0,2)
  {
    size_t tot = 4ull * 16 * 46 * 46 * 34;
    convt3d_kernel<<<DIV_UP(tot, 256), 256, 0, stream>>>(
        t3a, ct3_w, ct3_b, t3b, 32, 22, 22, 18, 16, 46, 46, 34, 0, 0, 2, 0);
  }
  bn3d(t3b, bn3_g, bn3_b, 16, 46 * 46 * 34, 1);
  {
    size_t tot = 4ull * 16 * 71944;
    conv3d1_kernel<<<DIV_UP(tot, 256), 256, 0, stream>>>(t3b, c3_w, c3_b, t3a,
                                                         16, 16, 71944);
  }
  bn3d(t3a, bnc3_g, bnc3_b, 16, 71944, 1);

  // ct4: 16->1, (46,46,34) -> (94,94,66), pad (0,0,2), fused sigmoid
  {
    size_t tot = 4ull * 94 * 94 * 66;
    convt3d_kernel<<<DIV_UP(tot, 256), 256, 0, stream>>>(
        t3a, ct4_w, ct4_b, t3b, 16, 46, 46, 34, 1, 94, 94, 66, 0, 0, 2, 1);
  }

  // grid_sample -> (4,150,150,150)
  {
    size_t tot = 4ull * 150 * 150 * 150;
    grid_sample_kernel<<<DIV_UP(tot, 256), 256, 0, stream>>>(t3b, grid, vol,
                                                             94, 94, 66);
  }

  // final masked gather -> (4*262144, 2)
  {
    size_t tot = 4ull * 262144;
    gather_out_kernel<<<DIV_UP(tot, 256), 256, 0, stream>>>(xpts, vol, out);
  }
}
